// PNAOriginal_62225486185133
// MI455X (gfx1250) — matrix operand order repaired and checked by reading, not compiled, not run on hardware
//
#include <hip/hip_runtime.h>
#include <math.h>

#define DIM      128
#define EDIM     64
#define PRE_IN   320        // 2*DIM + EDIM
#define POST_IN  1664       // 13*DIM
#define PRE_KC   10         // 320/32
#define POST_KC  52         // 1664/32
#define MIX_KC   4          // 128/32
#define AVGD     3.5f
#define EPSV     1e-5f

#define PRE_PACK_ELEMS  (8 * PRE_KC  * 32 * 16)   // 40960 bf16 = 80KB
#define POST_PACK_ELEMS (8 * POST_KC * 32 * 16)
#define MIX_PACK_ELEMS  (8 * MIX_KC  * 32 * 16)

typedef __attribute__((ext_vector_type(16))) __bf16 v16bf;
typedef __attribute__((ext_vector_type(16))) float  v16f;
typedef __attribute__((ext_vector_type(8)))  float  v8f;
typedef __attribute__((ext_vector_type(4)))  unsigned int u32x4;
typedef __attribute__((ext_vector_type(8)))  int    i32x8;
typedef __attribute__((ext_vector_type(4)))  int    i32x4;

// ---------------- helpers ----------------

// float atomic max/min via ordered-integer trick (monotone bit patterns)
__device__ __forceinline__ void atomicMaxF(float* a, float v) {
  if (v >= 0.0f) atomicMax((int*)a, __float_as_int(v));
  else           atomicMin((unsigned int*)a, __float_as_uint(v));
}
__device__ __forceinline__ void atomicMinF(float* a, float v) {
  if (v >= 0.0f) atomicMin((int*)a, __float_as_int(v));
  else           atomicMax((unsigned int*)a, __float_as_uint(v));
}

// Build a 16x32 bf16 A-fragment row-slice for this lane from a contiguous f32 row.
// ISA layout: lanes 0-15 row M=lane, K = {0..7, 16..23}; lanes 16-31 K = {8..15, 24..31}.
// Hardware bf16 converts via vector fptrunc (v_cvt_pk_bf16_f32).
__device__ __forceinline__ v16bf load_a_frag(const float* base, int o0, float scale) {
  float4 f0 = *(const float4*)(base + o0);
  float4 f1 = *(const float4*)(base + o0 + 4);
  float4 f2 = *(const float4*)(base + o0 + 16);
  float4 f3 = *(const float4*)(base + o0 + 20);
  v16f t = { f0.x, f0.y, f0.z, f0.w, f1.x, f1.y, f1.z, f1.w,
             f2.x, f2.y, f2.z, f2.w, f3.x, f3.y, f3.z, f3.w };
  t = t * scale;
  return __builtin_convertvector(t, v16bf);
}

// Tensor Data Mover: 1-D DMA (global -> LDS) of nbytes (multiple of 8).
// D# per CDNA5 ISA sec.8: group0 = {count=1, lds_addr,
// global_addr[56:0], type=2}; group1 = {data_size=8B, tensor_dim0=tile_dim0=N elems}.
__device__ __forceinline__ void tdm_load_1d(unsigned lds_addr, const void* gptr, unsigned nbytes) {
  unsigned long long ga = (unsigned long long)(size_t)gptr;
  unsigned elems = nbytes >> 3;                       // 8-byte elements
  u32x4 g0;
  g0[0] = 1u;                                          // count=1
  g0[1] = lds_addr;                                    // LDS byte address
  g0[2] = (unsigned)(ga & 0xFFFFFFFFu);                // global_addr[31:0]
  g0[3] = (unsigned)((ga >> 32) & 0x1FFFFFFu) | (2u << 30);  // addr[56:32] | type=2
  i32x8 g1;
  g1[0] = (3 << 16);                                   // data_size=3 (8B), wg_mask=0
  g1[1] = (int)((elems & 0xFFFFu) << 16);              // tensor_dim0[15:0]
  g1[2] = (int)(((elems >> 16) & 0xFFFFu) | (1u << 16)); // tensor_dim0[31:16] | tensor_dim1 lo
  g1[3] = (int)((elems & 0xFFFFu) << 16);              // tile_dim0
  g1[4] = 0;                                           // tile_dim1=0 (1-D), tile_dim2=0
  g1[5] = (int)elems;                                  // tensor_dim0_stride[31:0]
  g1[6] = 0;
  g1[7] = 0;
  i32x4 z4 = {0, 0, 0, 0};
#if __clang_major__ >= 23
  i32x8 z8 = {0, 0, 0, 0, 0, 0, 0, 0};
  __builtin_amdgcn_tensor_load_to_lds(g0, g1, z4, z4, z8, 0);
#else
  __builtin_amdgcn_tensor_load_to_lds(g0, g1, z4, z4, 0);
#endif
}

// ---------------- small elementwise kernels ----------------

__global__ void pna_fill_f32(float* p, float v, long n) {
  long i = (long)blockIdx.x * blockDim.x + threadIdx.x;
  if (i < n) p[i] = v;
}

__global__ void pna_degree(const int* __restrict__ dst, float* __restrict__ deg, long e) {
  long i = (long)blockIdx.x * blockDim.x + threadIdx.x;
  if (i < e) atomicAdd(&deg[dst[i]], 1.0f);
}

__global__ void pna_scalers(const float* __restrict__ deg,
                            float* __restrict__ amp, float* __restrict__ att, long n) {
  long i = (long)blockIdx.x * blockDim.x + threadIdx.x;
  if (i >= n) return;
  float d = deg[i];
  float logd = logf(d + 1.0f);
  amp[i] = logd / AVGD;
  att[i] = AVGD / fmaxf(logd, EPSV);
}

__global__ void pna_init_agg(float* S, float* S2, float* MX, float* MN, long n) {
  long i = (long)blockIdx.x * blockDim.x + threadIdx.x;
  if (i >= n) return;
  S[i] = 0.0f; S2[i] = 0.0f;
  MX[i] = -__builtin_inff();
  MN[i] =  __builtin_inff();
}

__global__ void pna_finalize(const float* __restrict__ deg,
                             float* S, float* S2, float* MX, float* MN, long n) {
  long i = (long)blockIdx.x * blockDim.x + threadIdx.x;
  if (i >= n) return;
  long node = i >> 7;          // /DIM
  float d = deg[node];
  if (d <= 0.0f) {
    S[i] = 0.0f; S2[i] = 0.0f; MX[i] = 0.0f; MN[i] = 0.0f;
  } else {
    float mean = S[i] / d;
    float var  = fmaxf(S2[i] / d - mean * mean, 0.0f);
    S[i]  = mean;
    S2[i] = sqrtf(var + EPSV);
  }
}

// Repack W [K x 128] (f32, row-major) into per-wave WMMA B-fragments (bf16).
// B layout (32x16): lanes 0-15 hold K={0..7,16..23} for column N=lane, lanes 16-31 K={8..15,24..31}.
__global__ void pna_pack_w(const float* __restrict__ W, int KC, __bf16* __restrict__ dst) {
  int tid = blockIdx.x * blockDim.x + threadIdx.x;
  int total = 8 * KC * 32;
  if (tid >= total) return;
  int lane = tid & 31;
  int kc   = (tid >> 5) % KC;
  int nt   = tid / (32 * KC);
  int n    = nt * 16 + (lane & 15);
  int kb   = kc * 32 + ((lane >> 4) ? 8 : 0);
  v16bf v;
  #pragma unroll
  for (int i = 0; i < 16; ++i)
    v[i] = (__bf16)W[(size_t)(kb + i + (i >= 8 ? 8 : 0)) * DIM + n];
  *(v16bf*)(dst + (size_t)tid * 16) = v;
}

// ---------------- edge GEMM + fused segment scatter ----------------
// One wave per 16-edge tile. K = 320 = [h_src(128) | h_dst(128) | e(64)].
// B fragments staged once per block into LDS via the Tensor Data Mover.
// Epilogue: 4 float atomics per output into L2-resident node buffers.

__global__ __launch_bounds__(128) void pna_edge_gemm_scatter(
    const float* __restrict__ hcur, const float* __restrict__ e,
    const int* __restrict__ src, const int* __restrict__ dst,
    const __bf16* __restrict__ wp, const float* __restrict__ bias,
    float* __restrict__ S, float* __restrict__ S2,
    float* __restrict__ MX, float* __restrict__ MN, long ntiles) {
  __shared__ __attribute__((aligned(32))) __bf16 sB[PRE_PACK_ELEMS];  // 80KB

  const int lane = threadIdx.x & 31;
  const int wave = threadIdx.x >> 5;
  const long tile = (long)blockIdx.x * 4 + wave;

  // Stage the whole packed pre-W for this layer into LDS (one TDM op).
  if (threadIdx.x == 0) {
    tdm_load_1d((unsigned)(size_t)(void*)sB, (const void*)wp, PRE_PACK_ELEMS * 2);
    __builtin_amdgcn_s_wait_tensorcnt(0);
  }
  __syncthreads();
  if (tile >= ntiles) return;                // wave-uniform: EXEC stays all-ones

  const long ebase = tile * 16;
  const int mrow = lane & 15;
  const int hi   = lane >> 4;
  const int o0   = hi * 8;

  const long eidx = ebase + mrow;
  const float* hs = hcur + (size_t)src[eidx] * DIM;
  const float* hd = hcur + (size_t)dst[eidx] * DIM;
  const float* er = e    + (size_t)eidx * EDIM;

  v8f acc[8];
  #pragma unroll
  for (int nt = 0; nt < 8; ++nt) acc[nt] = (v8f){0,0,0,0,0,0,0,0};

  #pragma unroll
  for (int kc = 0; kc < PRE_KC; ++kc) {
    const int kb = kc * 32;
    const float* base = (kb < 128) ? (hs + kb)
                      : (kb < 256) ? (hd + (kb - 128))
                                   : (er + (kb - 256));
    v16bf a = load_a_frag(base, o0, 1.0f);
    #pragma unroll
    for (int nt = 0; nt < 8; ++nt) {
      v16bf b = *(const v16bf*)(sB + ((size_t)(nt * PRE_KC + kc) * 32 + lane) * 16);
      acc[nt] = __builtin_amdgcn_wmma_f32_16x16x32_bf16(
          false, a, false, b, (short)0, acc[nt], false, false);
    }
  }

  // dst nodes for the 8 rows this lane owns (M = hi*8 + j)
  const int4* dp = (const int4*)(dst + ebase + o0);
  int4 dA = dp[0], dB = dp[1];
  int drow[8] = {dA.x, dA.y, dA.z, dA.w, dB.x, dB.y, dB.z, dB.w};

  #pragma unroll
  for (int nt = 0; nt < 8; ++nt) {
    const int col = nt * 16 + mrow;
    const float bc = bias[col];
    #pragma unroll
    for (int j = 0; j < 8; ++j) {
      float v = acc[nt][j] + bc;
      size_t off = (size_t)drow[j] * DIM + col;
      atomicAdd(S  + off, v);
      atomicAdd(S2 + off, v * v);
      atomicMaxF(MX + off, v);
      atomicMinF(MN + off, v);
    }
  }
}

// ---------------- node post GEMM ----------------
// z = [h | mean | mx | mn | std | amp*(...) | att*(...)], K = 1664, per-region scale.

__global__ __launch_bounds__(128) void pna_post_gemm(
    const float* __restrict__ hcur,
    const float* __restrict__ MEAN, const float* __restrict__ MXb,
    const float* __restrict__ MNb,  const float* __restrict__ STD,
    const float* __restrict__ amp,  const float* __restrict__ att,
    const __bf16* __restrict__ wp,  const float* __restrict__ bias,
    float* __restrict__ HT, long nnodes, long ntiles) {
  const int lane = threadIdx.x & 31;
  const int wave = threadIdx.x >> 5;
  const long tile = (long)blockIdx.x * 4 + wave;
  if (tile >= ntiles) return;
  const int mrow = lane & 15;
  const int hi   = lane >> 4;
  const int o0   = hi * 8;

  long node = tile * 16 + mrow;
  if (node >= nnodes) node = nnodes - 1;     // clamp loads; stores bounds-checked
  const float* rowh = hcur + (size_t)node * DIM;
  const float* regs[4] = { MEAN + (size_t)node * DIM, MXb + (size_t)node * DIM,
                           MNb  + (size_t)node * DIM, STD + (size_t)node * DIM };
  const float ampv = amp[node], attv = att[node];

  v8f acc[8];
  #pragma unroll
  for (int nt = 0; nt < 8; ++nt) acc[nt] = (v8f){0,0,0,0,0,0,0,0};

  #pragma unroll
  for (int kc = 0; kc < POST_KC; ++kc) {
    const int kb = kc * 32;
    const float* base;
    float scale;
    if (kb < DIM) { base = rowh + kb; scale = 1.0f; }
    else {
      int q   = kb - DIM;
      int blk = q >> 9;          // 512-wide scaler blocks: id / amp / att
      int r   = q & 511;
      scale = (blk == 0) ? 1.0f : ((blk == 1) ? ampv : attv);
      base  = regs[r >> 7] + (r & 127);
    }
    v16bf a = load_a_frag(base, o0, scale);
    #pragma unroll
    for (int nt = 0; nt < 8; ++nt) {
      v16bf b = *(const v16bf*)(wp + ((size_t)(nt * POST_KC + kc) * 32 + lane) * 16);
      acc[nt] = __builtin_amdgcn_wmma_f32_16x16x32_bf16(
          false, a, false, b, (short)0, acc[nt], false, false);
    }
  }

  const long rbase = tile * 16 + hi * 8;
  #pragma unroll
  for (int nt = 0; nt < 8; ++nt) {
    const int col = nt * 16 + mrow;
    const float bc = bias[col];
    #pragma unroll
    for (int j = 0; j < 8; ++j) {
      long n2 = rbase + j;
      if (n2 < nnodes) HT[(size_t)n2 * DIM + col] = acc[nt][j] + bc;
    }
  }
}

// ---------------- mix GEMM + leaky-relu + residual ----------------

__global__ __launch_bounds__(128) void pna_mix_residual(
    const float* __restrict__ hcur, const float* __restrict__ HT,
    const __bf16* __restrict__ wp, const float* __restrict__ bias,
    float* __restrict__ hout, long nnodes, long ntiles) {
  const int lane = threadIdx.x & 31;
  const int wave = threadIdx.x >> 5;
  const long tile = (long)blockIdx.x * 4 + wave;
  if (tile >= ntiles) return;
  const int mrow = lane & 15;
  const int hi   = lane >> 4;
  const int o0   = hi * 8;

  long node = tile * 16 + mrow;
  if (node >= nnodes) node = nnodes - 1;
  const float* rowt = HT + (size_t)node * DIM;

  v8f acc[8];
  #pragma unroll
  for (int nt = 0; nt < 8; ++nt) acc[nt] = (v8f){0,0,0,0,0,0,0,0};

  #pragma unroll
  for (int kc = 0; kc < MIX_KC; ++kc) {
    v16bf a = load_a_frag(rowt + kc * 32, o0, 1.0f);
    #pragma unroll
    for (int nt = 0; nt < 8; ++nt) {
      v16bf b = *(const v16bf*)(wp + ((size_t)(nt * MIX_KC + kc) * 32 + lane) * 16);
      acc[nt] = __builtin_amdgcn_wmma_f32_16x16x32_bf16(
          false, a, false, b, (short)0, acc[nt], false, false);
    }
  }

  const long rbase = tile * 16 + hi * 8;
  #pragma unroll
  for (int nt = 0; nt < 8; ++nt) {
    const int col = nt * 16 + mrow;
    const float bc = bias[col];
    #pragma unroll
    for (int j = 0; j < 8; ++j) {
      long n2 = rbase + j;
      if (n2 < nnodes) {
        float v = acc[nt][j] + bc;
        v = (v > 0.0f) ? v : 0.01f * v;                 // leaky_relu
        hout[(size_t)n2 * DIM + col] = hcur[(size_t)n2 * DIM + col] + v;
      }
    }
  }
}

// ---------------- host launcher ----------------

extern "C" void kernel_launch(void* const* d_in, const int* in_sizes, int n_in,
                              void* d_out, int out_size, void* d_ws, size_t ws_size,
                              hipStream_t stream) {
  const float* h0    = (const float*)d_in[0];
  const float* e     = (const float*)d_in[1];
  const int*   src   = (const int*)  d_in[2];
  const int*   dst   = (const int*)  d_in[3];
  const float* preW  = (const float*)d_in[4];
  const float* preB  = (const float*)d_in[5];
  const float* postW = (const float*)d_in[6];
  const float* postB = (const float*)d_in[7];
  const float* mixW  = (const float*)d_in[8];
  const float* mixB  = (const float*)d_in[9];

  const long N = (long)in_sizes[0] / DIM;
  const long E = (long)in_sizes[2];
  const int  L = in_sizes[4] / (PRE_IN * DIM);

  // workspace carve-out (256B aligned)
  char* w = (char*)d_ws;
  auto carve = [&](size_t bytes) -> char* {
    char* p = w;
    w += (bytes + 255) & ~(size_t)255;
    return p;
  };
  float* deg = (float*)carve((size_t)N * 4);
  float* amp = (float*)carve((size_t)N * 4);
  float* att = (float*)carve((size_t)N * 4);
  const size_t nd = (size_t)N * DIM * 4;
  float* S   = (float*)carve(nd);
  float* S2  = (float*)carve(nd);
  float* MX  = (float*)carve(nd);
  float* MN  = (float*)carve(nd);
  float* HT  = (float*)carve(nd);
  float* HA  = (float*)carve(nd);
  float* HB  = (float*)carve(nd);
  __bf16* packPre  = (__bf16*)carve((size_t)L * PRE_PACK_ELEMS  * 2);
  __bf16* packPost = (__bf16*)carve((size_t)L * POST_PACK_ELEMS * 2);
  __bf16* packMix  = (__bf16*)carve((size_t)L * MIX_PACK_ELEMS  * 2);

  const int thr = 256;
  const long ND = N * DIM;

  pna_fill_f32<<<(int)((N + thr - 1) / thr), thr, 0, stream>>>(deg, 0.0f, N);
  pna_degree  <<<(int)((E + thr - 1) / thr), thr, 0, stream>>>(dst, deg, E);
  pna_scalers <<<(int)((N + thr - 1) / thr), thr, 0, stream>>>(deg, amp, att, N);

  for (int l = 0; l < L; ++l) {
    pna_pack_w<<<(8 * PRE_KC  * 32 + thr - 1) / thr, thr, 0, stream>>>(
        preW  + (size_t)l * PRE_IN  * DIM, PRE_KC,  packPre  + (size_t)l * PRE_PACK_ELEMS);
    pna_pack_w<<<(8 * POST_KC * 32 + thr - 1) / thr, thr, 0, stream>>>(
        postW + (size_t)l * POST_IN * DIM, POST_KC, packPost + (size_t)l * POST_PACK_ELEMS);
    pna_pack_w<<<(8 * MIX_KC  * 32 + thr - 1) / thr, thr, 0, stream>>>(
        mixW  + (size_t)l * DIM    * DIM, MIX_KC,  packMix  + (size_t)l * MIX_PACK_ELEMS);
  }

  const long ET = E / 16;             // edge tiles
  const long NT = (N + 15) / 16;      // node tiles
  const float* hcur = h0;

  for (int l = 0; l < L; ++l) {
    pna_init_agg<<<(int)((ND + thr - 1) / thr), thr, 0, stream>>>(S, S2, MX, MN, ND);

    pna_edge_gemm_scatter<<<(int)((ET + 3) / 4), 128, 0, stream>>>(
        hcur, e, src, dst,
        packPre + (size_t)l * PRE_PACK_ELEMS, preB + (size_t)l * DIM,
        S, S2, MX, MN, ET);

    pna_finalize<<<(int)((ND + thr - 1) / thr), thr, 0, stream>>>(deg, S, S2, MX, MN, ND);

    pna_post_gemm<<<(int)((NT + 3) / 4), 128, 0, stream>>>(
        hcur, S, MX, MN, S2, amp, att,
        packPost + (size_t)l * POST_PACK_ELEMS, postB + (size_t)l * DIM,
        HT, N, NT);

    float* hout = (l == L - 1) ? (float*)d_out : ((l & 1) ? HB : HA);
    pna_mix_residual<<<(int)((NT + 3) / 4), 128, 0, stream>>>(
        hcur, HT,
        packMix + (size_t)l * MIX_PACK_ELEMS, mixB + (size_t)l * DIM,
        hout, N, NT);
    hcur = hout;
  }
}
